// AttentionLS_51419348468379
// MI455X (gfx1250) — matrix operand order repaired and checked by reading, not compiled, not run on hardware
//
#include <hip/hip_runtime.h>
#include <hip/hip_bf16.h>
#include <math.h>

// ---- problem constants (match reference) ----
#define NUM_HEAD 8
#define HEAD_DIM 64
#define LMK 32
#define WIN 128
#define DIM 512
#define EXT 64
#define TILE_K 256          // window keys per 128-query group
#define NKEY (LMK + TILE_K) // 288 total keys per query
#define SEQ 8192
#define BATCH 2
#define QKSCALE 0.125f      // 1/sqrt(64)

typedef _Float16 h16;
typedef __attribute__((ext_vector_type(16))) _Float16 v16h;
typedef __attribute__((ext_vector_type(8)))  _Float16 v8h;
typedef __attribute__((ext_vector_type(8)))  float    v8f;

union AB { v16h v; v8h h[2]; };

__device__ __forceinline__ v8f wmma_f16(v16h a, v16h b, v8f c) {
  // v_wmma_f32_16x16x32_f16  D = A(16x32) * B(32x16) + C
  return __builtin_amdgcn_wmma_f32_16x16x32_f16(false, a, false, b,
                                                (short)0, c, false, false);
}

// ---------------- conversion kernels ----------------
__global__ void __launch_bounds__(256) k_cvt_f16(const float* __restrict__ src,
                                                 h16* __restrict__ dst, int n) {
  int i = blockIdx.x * 256 + threadIdx.x;
  if (i < n) dst[i] = (h16)src[i];
}

// w: [K][N] row-major f32  ->  wt: [N][K] row-major f16
__global__ void __launch_bounds__(256) k_transpose_cvt(const float* __restrict__ w,
                                                       h16* __restrict__ wt,
                                                       int K, int N) {
  int i = blockIdx.x * 256 + threadIdx.x;
  if (i < K * N) {
    int k = i / N, n = i % N;
    wt[(long)n * K + k] = (h16)w[i];
  }
}

// ---------------- generic WMMA GEMM ----------------
// C[M x N] = A[M x 512] * Bt[N x 512]^T + bias, *scale
// grid: (M/128, N/64), block 256 (8 waves); wave w -> rows 16w..16w+15
__global__ void __launch_bounds__(256)
k_gemm(const h16* __restrict__ A, const h16* __restrict__ Bt,
       const float* __restrict__ bias, float scale, int N,
       float* __restrict__ out_f32, h16* __restrict__ out_f16) {
  int lane = threadIdx.x & 31, wave = threadIdx.x >> 5;
  int lo = lane & 15, hi = lane >> 4;
  long row0 = (long)blockIdx.x * 128 + wave * 16;
  int n0 = blockIdx.y * 64;
  const h16* arow = A + (row0 + lo) * DIM;   // A fragment: m = lo
  v8f acc[4] = {};
#pragma unroll
  for (int k0 = 0; k0 < DIM; k0 += 32) {
    __builtin_prefetch(arow + k0 + 64, 0, 1);
    AB a;                                    // A 16x32: K = hi*16+j
    a.h[0] = *(const v8h*)(arow + k0 + hi * 16);
    a.h[1] = *(const v8h*)(arow + k0 + hi * 16 + 8);
#pragma unroll
    for (int t = 0; t < 4; ++t) {
      const h16* brow = Bt + (long)(n0 + t * 16 + lo) * DIM;
      v16h b = *(const v16h*)(brow + k0 + hi * 16);  // B 32x16: n=lo, K=hi*16+j
      acc[t] = wmma_f16(a.v, b, acc[t]);
    }
  }
#pragma unroll
  for (int t = 0; t < 4; ++t) {
    int n = n0 + t * 16 + lo;
    float bn = bias ? bias[n] : 0.f;
#pragma unroll
    for (int i = 0; i < 8; ++i) {
      long m = row0 + i + hi * 8;
      float v = (acc[t][i] + bn) * scale;
      if (out_f32) out_f32[m * N + n] = v;
      if (out_f16) out_f16[m * N + n] = (h16)v;
    }
  }
}

// ---------------- row layernorm (f32 in -> f16 out) ----------------
__global__ void __launch_bounds__(256)
k_layernorm(const float* __restrict__ x, const float* __restrict__ g,
            const float* __restrict__ bb, h16* __restrict__ out, int ncols) {
  long row = blockIdx.x;
  const float* xr = x + row * ncols;
  __shared__ float red[8];
  int tid = threadIdx.x;
  float s = 0.f;
  for (int c = tid; c < ncols; c += 256) s += xr[c];
  for (int m = 16; m >= 1; m >>= 1) s += __shfl_xor(s, m, 32);
  if ((tid & 31) == 0) red[tid >> 5] = s;
  __syncthreads();
  float tot = 0.f;
  for (int i = 0; i < 8; ++i) tot += red[i];
  float mean = tot / ncols;
  __syncthreads();
  float v = 0.f;
  for (int c = tid; c < ncols; c += 256) { float d = xr[c] - mean; v += d * d; }
  for (int m = 16; m >= 1; m >>= 1) v += __shfl_xor(v, m, 32);
  if ((tid & 31) == 0) red[tid >> 5] = v;
  __syncthreads();
  float vt = 0.f;
  for (int i = 0; i < 8; ++i) vt += red[i];
  float rstd = rsqrtf(vt / ncols + 1e-5f);
  for (int c = tid; c < ncols; c += 256)
    out[row * (long)ncols + c] = (h16)((xr[c] - mean) * rstd * g[c] + bb[c]);
}

// ---------------- column softmax for hs (over S per (b,col)) ----------------
__global__ void __launch_bounds__(256)
k_colsoftmax(const float* __restrict__ hs, const unsigned char* __restrict__ mask,
             h16* __restrict__ out) {
  int col = blockIdx.x, b = blockIdx.y;
  const float* base = hs + (long)b * SEQ * 256 + col;
  const unsigned char* mrow = mask + (long)b * SEQ;
  __shared__ float red[8];
  int tid = threadIdx.x;
  float mx = -__builtin_inff();
  for (int s = tid; s < SEQ; s += 256) {
    float v = mrow[s] ? base[(long)s * 256] : -__builtin_inff();
    mx = fmaxf(mx, v);
  }
  for (int m = 16; m >= 1; m >>= 1) mx = fmaxf(mx, __shfl_xor(mx, m, 32));
  if ((tid & 31) == 0) red[tid >> 5] = mx;
  __syncthreads();
  float gm = red[0];
  for (int i = 1; i < 8; ++i) gm = fmaxf(gm, red[i]);
  __syncthreads();
  float sum = 0.f;
  for (int s = tid; s < SEQ; s += 256)
    sum += mrow[s] ? __expf(base[(long)s * 256] - gm) : 0.f;
  for (int m = 16; m >= 1; m >>= 1) sum += __shfl_xor(sum, m, 32);
  if ((tid & 31) == 0) red[tid >> 5] = sum;
  __syncthreads();
  float gs = 0.f;
  for (int i = 0; i < 8; ++i) gs += red[i];
  float inv = 1.f / gs;
  h16* ob = out + (long)b * SEQ * 256 + col;
  for (int s = tid; s < SEQ; s += 256)
    ob[(long)s * 256] = (h16)(mrow[s] ? __expf(base[(long)s * 256] - gm) * inv : 0.f);
}

// ---------------- landmark GEMM: K_c/V_c = hs^T * K / V ----------------
// grid (H, B); per (b,h): [32 x 8192] * [8192 x 64], WMMA with LDS transpose
#define LM_HS_ST 40
#define LM_KV_ST 48
__global__ void __launch_bounds__(256)
k_landmark(const h16* __restrict__ hs, const h16* __restrict__ K,
           const h16* __restrict__ V, float* __restrict__ Kc,
           float* __restrict__ Vc) {
  int h = blockIdx.x, b = blockIdx.y;
  __shared__ __attribute__((aligned(64))) h16 hsT[LMK][LM_HS_ST];     // [l][s]
  __shared__ __attribute__((aligned(64))) h16 kT[HEAD_DIM][LM_KV_ST]; // [d][s]
  __shared__ __attribute__((aligned(64))) h16 vT[HEAD_DIM][LM_KV_ST];
  int tid = threadIdx.x, lane = tid & 31, wave = tid >> 5;
  int lo = lane & 15, hi = lane >> 4;
  int isV = wave >> 2, mt = wave & 1, nt2 = (wave >> 1) & 1;
  v8f acc[2] = {};
  for (int s0 = 0; s0 < SEQ; s0 += 32) {
    __syncthreads();
    for (int i = tid; i < 32 * 32; i += 256) {               // hs chunk transposed
      int s = i >> 5, l = i & 31;
      hsT[l][s] = hs[((long)b * SEQ + s0 + s) * 256 + h * LMK + l];
    }
    for (int i = tid; i < 32 * 64; i += 256) {               // K/V chunks transposed
      int s = i >> 6, d = i & 63;
      long off = ((long)b * SEQ + s0 + s) * DIM + h * HEAD_DIM + d;
      kT[d][s] = K[off];
      vT[d][s] = V[off];
    }
    __syncthreads();
    AB a;                                                    // A: m=landmark, K=s
    const h16* ar = &hsT[mt * 16 + lo][0];
    a.h[0] = *(const v8h*)(ar + hi * 16);
    a.h[1] = *(const v8h*)(ar + hi * 16 + 8);
#pragma unroll
    for (int t = 0; t < 2; ++t) {
      int d0 = nt2 * 32 + t * 16;
      const h16* br = isV ? &vT[d0 + lo][0] : &kT[d0 + lo][0];
      v16h bf = *(const v16h*)(br + hi * 16);                // B: K=s, n=d
      acc[t] = wmma_f16(a.v, bf, acc[t]);
    }
  }
  float* dst = isV ? Vc : Kc;
#pragma unroll
  for (int t = 0; t < 2; ++t) {
    int d = nt2 * 32 + t * 16 + lo;
#pragma unroll
    for (int i = 0; i < 8; ++i) {
      int l = mt * 16 + i + hi * 8;
      dst[((long)b * LMK + l) * DIM + h * HEAD_DIM + d] = acc[t][i];
    }
  }
}

// ---------------- fused compressed+windowed flash attention ----------------
// grid (G=64, H, B), block 256 (8 waves); wave -> 16 queries; 288 keys = 9 chunks
#define VA_ST 304
__global__ void __launch_bounds__(256)
k_attn(const h16* __restrict__ Q, const h16* __restrict__ K,
       const h16* __restrict__ V, const h16* __restrict__ Kc,
       const h16* __restrict__ Vc, const unsigned char* __restrict__ mask,
       h16* __restrict__ Co) {
  int g = blockIdx.x, h = blockIdx.y, b = blockIdx.z;
  __shared__ __attribute__((aligned(64))) h16 Vall[HEAD_DIM][VA_ST]; // [hd][key]
  __shared__ __attribute__((aligned(64))) h16 Pst[8][16][40];        // per-wave P
  int tid = threadIdx.x, lane = tid & 31, wave = tid >> 5;
  int lo = lane & 15, hi = lane >> 4;
  int s_base = g * WIN;
  // stage V_all = [V_c | V window (zero-padded / masked)] transposed to [hd][key]
  for (int i = tid; i < HEAD_DIM * NKEY; i += 256) {
    int d = i / NKEY, n = i % NKEY;
    h16 val = (h16)0.f;
    if (n < LMK) {
      val = Vc[((long)b * LMK + n) * DIM + h * HEAD_DIM + d];
    } else {
      int s = s_base - EXT + (n - LMK);
      if (s >= 0 && s < SEQ && mask[(long)b * SEQ + s])
        val = V[((long)b * SEQ + s) * DIM + h * HEAD_DIM + d];
    }
    Vall[d][n] = val;
  }
  __syncthreads();
  // Q fragments for this wave's 16 rows (2 k-steps over HEAD_DIM=64)
  const h16* qr = Q + ((long)b * SEQ + s_base + wave * 16 + lo) * DIM + h * HEAD_DIM;
  AB qa[2];
#pragma unroll
  for (int ks = 0; ks < 2; ++ks) {
    qa[ks].h[0] = *(const v8h*)(qr + ks * 32 + hi * 16);
    qa[ks].h[1] = *(const v8h*)(qr + ks * 32 + hi * 16 + 8);
  }
  v8f out[4] = {};
  float mrow[8], lrow[8];
#pragma unroll
  for (int i = 0; i < 8; ++i) { mrow[i] = -__builtin_inff(); lrow[i] = 0.f; }

  for (int c = 0; c < 9; ++c) {          // 9 chunks of 32 keys (chunk 0 = landmarks)
    v8f sv[2];
#pragma unroll
    for (int t = 0; t < 2; ++t) {        // two 16-key n-tiles per chunk
      int n = c * 32 + t * 16 + lo;      // this lane's key column
      v16h b0, b1;
      bool ok;
      if (n < LMK) {
        ok = true;
        const h16* kr = Kc + ((long)b * LMK + n) * DIM + h * HEAD_DIM;
        b0 = *(const v16h*)(kr + hi * 16);
        b1 = *(const v16h*)(kr + 32 + hi * 16);
      } else {
        int s = s_base - EXT + (n - LMK);
        int sc = s < 0 ? 0 : (s >= SEQ ? SEQ - 1 : s);
        ok = (s >= 0) && (s < SEQ) && (mask[(long)b * SEQ + sc] != 0);
        const h16* kr = K + ((long)b * SEQ + sc) * DIM + h * HEAD_DIM;
        b0 = *(const v16h*)(kr + hi * 16);
        b1 = *(const v16h*)(kr + 32 + hi * 16);
      }
      if (!ok) {
        for (int i = 0; i < 16; ++i) { b0[i] = (h16)0.f; b1[i] = (h16)0.f; }
      }
      v8f sacc = {};
      sacc = wmma_f16(qa[0].v, b0, sacc);
      sacc = wmma_f16(qa[1].v, b1, sacc);
      if (!ok)
        for (int i = 0; i < 8; ++i) sacc[i] = -__builtin_inff();
      sv[t] = sacc;
    }
    // online softmax update (row = (vgpr i, lane-group))
#pragma unroll
    for (int i = 0; i < 8; ++i) {
      float m2 = fmaxf(sv[0][i], sv[1][i]);
      for (int msk = 8; msk >= 1; msk >>= 1) m2 = fmaxf(m2, __shfl_xor(m2, msk, 32));
      float mn = fmaxf(mrow[i], m2);
      float rs = (mrow[i] == -__builtin_inff()) ? 0.f : __expf(mrow[i] - mn);
#pragma unroll
      for (int t = 0; t < 4; ++t) out[t][i] *= rs;
      float p0 = (sv[0][i] == -__builtin_inff()) ? 0.f : __expf(sv[0][i] - mn);
      float p1 = (sv[1][i] == -__builtin_inff()) ? 0.f : __expf(sv[1][i] - mn);
      float ps = p0 + p1;
      for (int msk = 8; msk >= 1; msk >>= 1) ps += __shfl_xor(ps, msk, 32);
      mrow[i] = mn;
      lrow[i] = lrow[i] * rs + ps;
      sv[0][i] = p0;
      sv[1][i] = p1;
    }
    // stage P (C-layout -> A-layout) through per-wave LDS
#pragma unroll
    for (int t = 0; t < 2; ++t)
#pragma unroll
      for (int i = 0; i < 8; ++i)
        Pst[wave][i + hi * 8][t * 16 + lo] = (h16)sv[t][i];
    AB pa;
    const h16* pr = &Pst[wave][lo][0];
    pa.h[0] = *(const v8h*)(pr + hi * 16);
    pa.h[1] = *(const v8h*)(pr + hi * 16 + 8);
#pragma unroll
    for (int t = 0; t < 4; ++t) {        // P (16x32) * V_all chunk (32x16) x4 hd-tiles
      const h16* vr = &Vall[t * 16 + lo][0];
      v16h bf = *(const v16h*)(vr + c * 32 + hi * 16);
      out[t] = wmma_f16(pa.v, bf, out[t]);
    }
  }
  // normalize and store (padded query rows -> 0)
#pragma unroll
  for (int t = 0; t < 4; ++t) {
#pragma unroll
    for (int i = 0; i < 8; ++i) {
      int m = i + hi * 8;
      long sq = (long)s_base + wave * 16 + m;
      float inv = (lrow[i] > 0.f) ? 1.f / lrow[i] : 0.f;
      float val = out[t][i] * inv;
      if (!mask[(long)b * SEQ + sq]) val = 0.f;
      Co[((long)b * SEQ + sq) * DIM + h * HEAD_DIM + t * 16 + lo] = (h16)val;
    }
  }
}

// ---------------- host orchestration ----------------
extern "C" void kernel_launch(void* const* d_in, const int* in_sizes, int n_in,
                              void* d_out, int out_size, void* d_ws, size_t ws_size,
                              hipStream_t stream) {
  (void)in_sizes; (void)n_in; (void)out_size; (void)ws_size;
  const float* X       = (const float*)d_in[0];
  const unsigned char* mask = (const unsigned char*)d_in[1];
  const float* Wq_w = (const float*)d_in[2];
  const float* Wq_b = (const float*)d_in[3];
  const float* Wk_w = (const float*)d_in[4];
  const float* Wk_b = (const float*)d_in[5];
  const float* Wv_w = (const float*)d_in[6];
  const float* Wv_b = (const float*)d_in[7];
  const float* Wo_w = (const float*)d_in[8];
  const float* Wo_b = (const float*)d_in[9];
  const float* ln_l_g = (const float*)d_in[10];
  const float* ln_l_b = (const float*)d_in[11];
  const float* ln_s_g = (const float*)d_in[12];
  const float* ln_s_b = (const float*)d_in[13];
  const float* dconv_w = (const float*)d_in[14];
  const float* dconv_b = (const float*)d_in[15];

  const long M = (long)BATCH * SEQ;  // 16384 rows
  char* base = (char*)d_ws;
  size_t off = 0;
  auto alloc = [&](size_t bytes) {
    char* p = base + off;
    off = (off + bytes + 255) & ~(size_t)255;
    return p;
  };
  h16*   Xh   = (h16*)alloc((size_t)M * DIM * 2);
  h16*   WqT  = (h16*)alloc((size_t)DIM * DIM * 2);
  h16*   WkT  = (h16*)alloc((size_t)DIM * DIM * 2);
  h16*   WvT  = (h16*)alloc((size_t)DIM * DIM * 2);
  h16*   WoT  = (h16*)alloc((size_t)DIM * DIM * 2);
  h16*   WdT  = (h16*)alloc((size_t)256 * DIM * 2);
  h16*   Qh   = (h16*)alloc((size_t)M * DIM * 2);
  h16*   Kh   = (h16*)alloc((size_t)M * DIM * 2);
  h16*   Vh   = (h16*)alloc((size_t)M * DIM * 2);
  h16*   HSh  = (h16*)alloc((size_t)M * 256 * 2);
  float* T32  = (float*)alloc((size_t)M * DIM * 4);
  float* Kc32 = (float*)alloc((size_t)BATCH * LMK * DIM * 4);
  float* Vc32 = (float*)alloc((size_t)BATCH * LMK * DIM * 4);
  h16*   Kch  = (h16*)alloc((size_t)BATCH * LMK * DIM * 2);
  h16*   Vch  = (h16*)alloc((size_t)BATCH * LMK * DIM * 2);
  h16*   Ch   = (h16*)alloc((size_t)M * DIM * 2);

  // 1) precision conversions (X) and one-time weight transposes
  {
    int n = (int)(M * DIM);
    k_cvt_f16<<<dim3((n + 255) / 256), 256, 0, stream>>>(X, Xh, n);
  }
  k_transpose_cvt<<<dim3((DIM * DIM + 255) / 256), 256, 0, stream>>>(Wq_w, WqT, DIM, DIM);
  k_transpose_cvt<<<dim3((DIM * DIM + 255) / 256), 256, 0, stream>>>(Wk_w, WkT, DIM, DIM);
  k_transpose_cvt<<<dim3((DIM * DIM + 255) / 256), 256, 0, stream>>>(Wv_w, WvT, DIM, DIM);
  k_transpose_cvt<<<dim3((DIM * DIM + 255) / 256), 256, 0, stream>>>(Wo_w, WoT, DIM, DIM);
  k_transpose_cvt<<<dim3((DIM * 256 + 255) / 256), 256, 0, stream>>>(dconv_w, WdT, DIM, 256);

  // 2) hs = X @ dconv + b  (f32), then column softmax over S -> f16
  k_gemm<<<dim3(128, 4), 256, 0, stream>>>(Xh, WdT, dconv_b, 1.f, 256, T32, nullptr);
  k_colsoftmax<<<dim3(256, 2), 256, 0, stream>>>(T32, mask, HSh);

  // 3) Q (scaled, f16 direct), K/V (f32 -> layernorm -> f16)
  k_gemm<<<dim3(128, 8), 256, 0, stream>>>(Xh, WqT, Wq_b, QKSCALE, DIM, nullptr, Qh);
  k_gemm<<<dim3(128, 8), 256, 0, stream>>>(Xh, WkT, Wk_b, 1.f, DIM, T32, nullptr);
  k_layernorm<<<dim3((unsigned)M), 256, 0, stream>>>(T32, ln_l_g, ln_l_b, Kh, DIM);
  k_gemm<<<dim3(128, 8), 256, 0, stream>>>(Xh, WvT, Wv_b, 1.f, DIM, T32, nullptr);
  k_layernorm<<<dim3((unsigned)M), 256, 0, stream>>>(T32, ln_l_g, ln_l_b, Vh, DIM);

  // 4) landmarks: K_c/V_c = hs^T K / hs^T V, then layernorm -> f16
  k_landmark<<<dim3(NUM_HEAD, BATCH), 256, 0, stream>>>(HSh, Kh, Vh, Kc32, Vc32);
  k_layernorm<<<dim3(BATCH * LMK), 256, 0, stream>>>(Kc32, ln_s_g, ln_s_b, Kch, DIM);
  k_layernorm<<<dim3(BATCH * LMK), 256, 0, stream>>>(Vc32, ln_s_g, ln_s_b, Vch, DIM);

  // 5) fused compressed+windowed attention -> C (f16)
  k_attn<<<dim3(SEQ / WIN, NUM_HEAD, BATCH), 256, 0, stream>>>(Qh, Kh, Vh, Kch, Vch,
                                                               mask, Ch);

  // 6) out = C @ Wo + b  (f32 into d_out)
  k_gemm<<<dim3(128, 8), 256, 0, stream>>>(Ch, WoT, Wo_b, 1.f, DIM,
                                           (float*)d_out, nullptr);
}
